// EEGRCformer_62680752718154
// MI455X (gfx1250) — compile-verified, matrix operand order checked by reading, not run on hardware
//
#include <hip/hip_runtime.h>
#include <hip/hip_bf16.h>
#include <math.h>

// ---------------- problem constants ----------------
#define BATCH 128
#define CH    62
#define LEN   2000
#define WINL  250
#define STRD  50
#define NW    36
#define NBINS 44     // DFT bins 1..44
#define NB    7      // bands
#define FEAT  504    // 36*14
#define FEATP 512    // padded K for WMMA
#define DM    128
#define NHD   4
#define DH    32
#define NLAY  3
#define NCL   5
#define NTOK  (BATCH*NCL)   // 640
#define EPSF  1e-9f
#define LNEPS 1e-5f

typedef __attribute__((ext_vector_type(16))) _Float16 v16h;
typedef __attribute__((ext_vector_type(8)))  float    v8f;

__constant__ int c_band_lo[NB] = {0,3,7,11,15,19,29};   // ps index = bin-1
__constant__ int c_band_n [NB] = {3,4,4,4,4,10,15};
__constant__ int c_sizes [NCL] = {13,13,12,12,12};

// K index covered by packed element e of a 16-bit A/B fragment (ISA 7.12.2):
// lanes 0-15: VGPR0..3 -> K 0..7, VGPR4..7 -> K 16..23; lanes 16-31: +8.
__device__ __forceinline__ int kmap(int e, int hi) {
  return (e < 8) ? (e + 8*hi) : (e + 8 + 8*hi);
}

// ---------------- DFT basis + Hann table ----------------
// basisT: [96][256] f16, rows 0..43 = cos(2*pi*bin*t/250), rows 44..87 = sin, rest 0
__global__ void basis_kernel(_Float16* __restrict__ basisT, float* __restrict__ hann) {
  const float twopi = 6.283185307179586f;
  for (int idx = blockIdx.x*blockDim.x + threadIdx.x; idx < 96*256; idx += gridDim.x*blockDim.x) {
    int n = idx >> 8, t = idx & 255;
    float v = 0.f;
    if (t < WINL) {
      if (n < NBINS)        { int ph = ((n+1)*t) % WINL;       v = cosf(twopi*(float)ph/(float)WINL); }
      else if (n < 2*NBINS) { int ph = ((n-NBINS+1)*t) % WINL; v = sinf(twopi*(float)ph/(float)WINL); }
    }
    basisT[idx] = (_Float16)v;
  }
  for (int t = blockIdx.x*blockDim.x + threadIdx.x; t < 256; t += gridDim.x*blockDim.x)
    hann[t] = (t < WINL) ? 0.5f*(1.0f - cosf(6.283185307179586f*(float)t/(float)WINL)) : 0.f;
}

// ---------------- feature extraction: windowed DFT as WMMA GEMM ----------------
// one block per (b,c); A = 48x256 f16 windowed signal (36 real rows), B = basisT 96x256
__global__ __launch_bounds__(256)
void feat_kernel(const float* __restrict__ x, const _Float16* __restrict__ basisT,
                 const float* __restrict__ hann,
                 float* __restrict__ featf32, _Float16* __restrict__ featf16)
{
  __shared__ float    xl[LEN];
  __shared__ _Float16 Awin[48*256];
  __shared__ float    Dst[48*96];
  __shared__ float    psd[NW*NB], de[NW*NB];
  __shared__ float    redA[256], redB[256];
  __shared__ float    stats[4];

  const int bc  = blockIdx.x;
  const int tid = threadIdx.x;

  for (int i = tid; i < LEN; i += 256) xl[i] = x[(size_t)bc*LEN + i];
  __syncthreads();

  for (int i = tid; i < 48*256; i += 256) {
    int w = i >> 8, t = i & 255;
    float v = 0.f;
    if (w < NW && t < WINL) v = xl[w*STRD + t] * hann[t];
    Awin[i] = (_Float16)v;
  }
  __syncthreads();

  const int wv = tid >> 5, lane = tid & 31;
  const int hi = lane >> 4, r = lane & 15;
  for (int tile = wv; tile < 18; tile += 8) {      // 3 M-tiles x 6 N-tiles
    const int tm = tile / 6, tn = tile % 6;
    v8f acc = {};
    for (int k0 = 0; k0 < 256; k0 += 32) {
      v16h a, bfr;
      #pragma unroll
      for (int j = 0; j < 8; ++j) {
        const int kk = kmap(2*j, hi);
        const int m = tm*16 + r, n = tn*16 + r;
        a[2*j]     = Awin[m*256 + k0 + kk];
        a[2*j+1]   = Awin[m*256 + k0 + kk + 1];
        bfr[2*j]   = basisT[n*256 + k0 + kk];
        bfr[2*j+1] = basisT[n*256 + k0 + kk + 1];
      }
      acc = __builtin_amdgcn_wmma_f32_16x16x32_f16(false, a, false, bfr, (short)0, acc, false, false);
    }
    #pragma unroll
    for (int i = 0; i < 8; ++i) {
      const int row = tm*16 + i + 8*hi;
      const int col = tn*16 + r;
      Dst[row*96 + col] = acc[i];
    }
  }
  __syncthreads();

  // band PSD + differential entropy
  for (int task = tid; task < NW*NB; task += 256) {
    const int w = task / NB, f = task % NB;
    const int lo = c_band_lo[f], n = c_band_n[f];
    float s = 0.f;
    for (int q = 0; q < n; ++q) {
      float re = Dst[w*96 + lo + q];
      float im = Dst[w*96 + NBINS + lo + q];
      s += (re*re + im*im) * (1.0f/(float)WINL);
    }
    s /= (float)n;
    psd[task] = s;
    de[task]  = 0.5f * logf(17.079468445347132f * s + EPSF);   // 2*pi*e
  }
  __syncthreads();

  // normalize psd (mean, ddof=1 std over 252 values)
  float v0 = (tid < NW*NB) ? psd[tid] : 0.f;
  redA[tid] = v0; redB[tid] = v0*v0;
  __syncthreads();
  for (int s = 128; s > 0; s >>= 1) { if (tid < s) { redA[tid]+=redA[tid+s]; redB[tid]+=redB[tid+s]; } __syncthreads(); }
  if (tid == 0) {
    const float n = (float)(NW*NB);
    float m = redA[0]/n, var = (redB[0] - n*m*m)/(n-1.f);
    stats[0] = m; stats[1] = sqrtf(fmaxf(var, 0.f));
  }
  __syncthreads();
  float v1 = (tid < NW*NB) ? de[tid] : 0.f;
  redA[tid] = v1; redB[tid] = v1*v1;
  __syncthreads();
  for (int s = 128; s > 0; s >>= 1) { if (tid < s) { redA[tid]+=redA[tid+s]; redB[tid]+=redB[tid+s]; } __syncthreads(); }
  if (tid == 0) {
    const float n = (float)(NW*NB);
    float m = redA[0]/n, var = (redB[0] - n*m*m)/(n-1.f);
    stats[2] = m; stats[3] = sqrtf(fmaxf(var, 0.f));
  }
  __syncthreads();

  const size_t b32 = (size_t)bc * FEAT;
  const size_t b16 = (size_t)bc * FEATP;
  for (int task = tid; task < NW*NB; task += 256) {
    const int w = task / NB, f = task % NB;
    float np = (psd[task] - stats[0]) / (stats[1] + EPSF);
    float nd = (de[task]  - stats[2]) / (stats[3] + EPSF);
    featf32[b32 + w*14 + f]     = np;
    featf32[b32 + w*14 + 7 + f] = nd;
    featf16[b16 + w*14 + f]     = (_Float16)np;
    featf16[b16 + w*14 + 7 + f] = (_Float16)nd;
  }
  for (int i = FEAT + tid; i < FEATP; i += 256) featf16[b16 + i] = (_Float16)0.f;
}

// ---------------- generic WMMA GEMM: Y[m][n] = act(sum_k A[m][k]*W[n][k] + bias[n]) ----------------
// one wave per 16x16 output tile; K multiple of 32; M,N multiples of 16
__global__ __launch_bounds__(32)
void wmma_gemm_kernel(const _Float16* __restrict__ A, int lda,
                      const _Float16* __restrict__ W, int ldw,
                      const float* __restrict__ bias,
                      float* __restrict__ out, _Float16* __restrict__ out16,
                      int M, int N, int K, int relu)
{
  (void)M;
  const int ntn  = N >> 4;
  const int tm   = blockIdx.x / ntn;
  const int tn   = blockIdx.x % ntn;
  const int lane = threadIdx.x & 31;
  const int hi   = lane >> 4, r = lane & 15;
  const int m = tm*16 + r;     // A row for this lane
  const int n = tn*16 + r;     // W row (= output column)
  v8f acc = {};
  for (int k0 = 0; k0 < K; k0 += 32) {
    v16h a, b;
    #pragma unroll
    for (int j = 0; j < 8; ++j) {
      const int kk = kmap(2*j, hi);
      a[2*j]   = A[(size_t)m*lda + k0 + kk];
      a[2*j+1] = A[(size_t)m*lda + k0 + kk + 1];
      b[2*j]   = W[(size_t)n*ldw + k0 + kk];
      b[2*j+1] = W[(size_t)n*ldw + k0 + kk + 1];
    }
    acc = __builtin_amdgcn_wmma_f32_16x16x32_f16(false, a, false, b, (short)0, acc, false, false);
  }
  const int col = tn*16 + r;
  const float bs = bias ? bias[col] : 0.f;
  #pragma unroll
  for (int i = 0; i < 8; ++i) {
    const int row = tm*16 + i + 8*hi;
    float v = acc[i] + bs;
    if (relu) v = fmaxf(v, 0.f);
    out[(size_t)row*N + col] = v;
    if (out16) out16[(size_t)row*N + col] = (_Float16)v;
  }
}

// ---------------- small conversion kernels ----------------
__global__ void f32_to_f16_kernel(const float* __restrict__ s, _Float16* __restrict__ d, int n) {
  int i = blockIdx.x*blockDim.x + threadIdx.x;
  if (i < n) d[i] = (_Float16)s[i];
}
__global__ void pad_projw_kernel(const float* __restrict__ s, _Float16* __restrict__ d) {
  int idx = blockIdx.x*blockDim.x + threadIdx.x;
  if (idx >= DM*FEATP) return;
  int n = idx / FEATP, k = idx % FEATP;
  d[idx] = (_Float16)((k < FEAT) ? s[n*FEAT + k] : 0.f);
}

// ---------------- LayerNorm (+optional residual, relu, f16 mirror), D = 128 ----------------
__global__ __launch_bounds__(DM)
void ln_kernel(const float* __restrict__ x, const float* __restrict__ res,
               const float* __restrict__ g, const float* __restrict__ b,
               float* __restrict__ out, _Float16* __restrict__ out16, int relu)
{
  __shared__ float red[DM];
  __shared__ float sm, sv;
  const int row = blockIdx.x, d = threadIdx.x;
  float v = x[(size_t)row*DM + d] + (res ? res[(size_t)row*DM + d] : 0.f);
  red[d] = v; __syncthreads();
  for (int s = 64; s > 0; s >>= 1) { if (d < s) red[d] += red[d+s]; __syncthreads(); }
  if (d == 0) sm = red[0] / (float)DM;
  __syncthreads();
  float c = v - sm;
  red[d] = c*c; __syncthreads();
  for (int s = 64; s > 0; s >>= 1) { if (d < s) red[d] += red[d+s]; __syncthreads(); }
  if (d == 0) sv = red[0] / (float)DM;
  __syncthreads();
  float y = c * rsqrtf(sv + LNEPS) * g[d] + b[d];
  if (relu) y = fmaxf(y, 0.f);
  out[(size_t)row*DM + d] = y;
  if (out16) out16[(size_t)row*DM + d] = (_Float16)y;
}

// ---------------- clustering ----------------
__device__ __forceinline__ float dist3(const float* a, const float* b) {
  float dx=a[0]-b[0], dy=a[1]-b[1], dz=a[2]-b[2];
  return sqrtf(fmaxf(dx*dx+dy*dy+dz*dz, 0.f));
}
// farthest-point sampling on a 62x62 distance matrix (single thread; matches jnp.argmax ties)
__device__ void fps62(const float* dist, int* sel) {
  float best = -1.f; int start = 0;
  for (int i = 0; i < CH; ++i) {
    float s = 0.f;
    for (int j = 0; j < CH; ++j) s += dist[i*CH + j];
    if (s > best) { best = s; start = i; }
  }
  float md[CH];
  for (int j = 0; j < CH; ++j) md[j] = dist[start*CH + j];
  sel[0] = start;
  for (int it = 1; it < NCL; ++it) {
    float mx = md[0]; int far = 0;
    for (int j = 1; j < CH; ++j) if (md[j] > mx) { mx = md[j]; far = j; }
    sel[it] = far;
    for (int j = 0; j < CH; ++j) md[j] = fminf(md[j], dist[far*CH + j]);
  }
}

__global__ void cluster_init_kernel(float* sums, float* cnts) {
  int i = threadIdx.x;
  if (i < NCL*3) sums[i] = 0.f;
  if (i < NCL)   cnts[i] = 0.f;
}

__global__ __launch_bounds__(64)
void centers0_kernel(const float* __restrict__ pos_emb, float* __restrict__ centers) {
  __shared__ float dist[CH*CH];
  __shared__ int sel[NCL];
  const int tid = threadIdx.x;
  for (int e = tid; e < CH*CH; e += 64) {
    int i = e / CH, j = e % CH;
    dist[e] = dist3(pos_emb + i*3, pos_emb + j*3);     // pos = pos_emb[0]
  }
  __syncthreads();
  if (tid == 0) fps62(dist, sel);
  __syncthreads();
  if (tid < NCL*3) centers[tid] = pos_emb[sel[tid/3]*3 + (tid%3)];
}

__global__ __launch_bounds__(256)
void cluster_batch_kernel(const float* __restrict__ featf32, const float* __restrict__ pos_emb,
                          float* __restrict__ sums, float* __restrict__ cnts)
{
  __shared__ float dist[CH*CH];
  __shared__ int sel[NCL];
  const int b = blockIdx.x, tid = threadIdx.x;
  const float* fb = featf32 + (size_t)b*CH*FEAT;
  for (int e = tid; e < CH*CH; e += 256) {
    int i = e / CH, j = e % CH;
    const float* fi = fb + (size_t)i*FEAT;
    const float* fj = fb + (size_t)j*FEAT;
    float s = 0.f;
    for (int k = 0; k < FEAT; ++k) { float d = fi[k]-fj[k]; s += d*d; }
    dist[e] = sqrtf(s);
  }
  __syncthreads();
  if (tid == 0) fps62(dist, sel);
  __syncthreads();
  if (tid < CH) {
    const float* pb = pos_emb + (size_t)b*CH*3;
    float bd = 1e30f; int ba = 0;
    for (int k = 0; k < NCL; ++k) {
      float d = dist3(pb + tid*3, pb + sel[k]*3);
      if (d < bd) { bd = d; ba = k; }
    }
    atomicAdd(&sums[ba*3+0], pb[tid*3+0]);
    atomicAdd(&sums[ba*3+1], pb[tid*3+1]);
    atomicAdd(&sums[ba*3+2], pb[tid*3+2]);
    atomicAdd(&cnts[ba], 1.f);
  }
}

__global__ __launch_bounds__(64)
void assign_kernel(const float* __restrict__ pos_emb, const float* __restrict__ sums,
                   const float* __restrict__ cnts, const float* __restrict__ centers,
                   int* __restrict__ assign)
{
  __shared__ float avg[NCL*3], cnew[NCL*3];
  __shared__ int order[CH*NCL];
  const int tid = threadIdx.x;
  if (tid < NCL*3) {
    float cn = cnts[tid/3];
    avg[tid] = (cn > 0.f) ? sums[tid] / fmaxf(cn, 1.f) : 0.f;
  }
  __syncthreads();
  if (tid < NCL) {
    float bd = 1e30f; int bj = 0;
    for (int j = 0; j < NCL; ++j) {
      float d = dist3(centers + tid*3, avg + j*3);
      if (d < bd) { bd = d; bj = j; }
    }
    for (int d = 0; d < 3; ++d)
      cnew[tid*3+d] = 0.8f*centers[tid*3+d] + 0.2f*avg[bj*3+d];
  }
  __syncthreads();
  if (tid < CH) {
    float dd[NCL];
    for (int k = 0; k < NCL; ++k) dd[k] = dist3(pos_emb + tid*3, cnew + k*3);
    int ord[NCL] = {0,1,2,3,4};                  // stable insertion sort (== jnp.argsort)
    for (int a2 = 1; a2 < NCL; ++a2) {
      int key = ord[a2]; float kd = dd[key]; int p = a2-1;
      while (p >= 0 && dd[ord[p]] > kd) { ord[p+1] = ord[p]; --p; }
      ord[p+1] = key;
    }
    for (int k = 0; k < NCL; ++k) order[tid*NCL + k] = ord[k];
  }
  __syncthreads();
  if (tid == 0) {                                // greedy capacity-constrained scan
    int counts[NCL] = {0,0,0,0,0};
    for (int c = 0; c < CH; ++c) {
      int sl = 0;
      for (int j = 0; j < NCL; ++j) {
        int cd = order[c*NCL + j];
        if (counts[cd] < c_sizes[cd]) { sl = j; break; }
      }
      int cl = order[c*NCL + sl];
      counts[cl]++; assign[c] = cl;
    }
  }
}

// ---------------- token pooling + positional encoding ----------------
__global__ __launch_bounds__(DM)
void pool_kernel(const float* __restrict__ pf, const int* __restrict__ assign,
                 const float* __restrict__ pos_enc,
                 float* __restrict__ h, _Float16* __restrict__ h16)
{
  const int b = blockIdx.x / NCL, k = blockIdx.x % NCL, d = threadIdx.x;
  float s = 0.f;
  for (int c = 0; c < CH; ++c)
    if (assign[c] == k) s += pf[((size_t)b*CH + c)*DM + d];
  float v = s / (float)c_sizes[k] + pos_enc[k*DM + d];
  h[((size_t)b*NCL + k)*DM + d]   = v;
  h16[((size_t)b*NCL + k)*DM + d] = (_Float16)v;
}

// ---------------- attention (T=5, per (batch, head) wave) ----------------
__global__ __launch_bounds__(32)
void attn_kernel(const float* __restrict__ qkv, float* __restrict__ attno,
                 _Float16* __restrict__ attno16)
{
  __shared__ float q[NCL][DH], kk[NCL][DH], vv[NCL][DH], sc[NCL][NCL];
  const int b = blockIdx.x / NHD, hh = blockIdx.x % NHD;
  const int d = threadIdx.x;
  for (int t = 0; t < NCL; ++t) {
    size_t base = ((size_t)b*NCL + t)*(3*DM) + hh*DH + d;
    q[t][d]  = qkv[base];
    kk[t][d] = qkv[base + DM];
    vv[t][d] = qkv[base + 2*DM];
  }
  __syncthreads();
  if (d < NCL*NCL) {
    int i = d / NCL, j = d % NCL;
    float s = 0.f;
    for (int e = 0; e < DH; ++e) s += q[i][e]*kk[j][e];
    sc[i][j] = s * 0.17677669529663687f;          // 1/sqrt(32)
  }
  __syncthreads();
  if (d < NCL) {
    float mx = sc[d][0];
    for (int j = 1; j < NCL; ++j) mx = fmaxf(mx, sc[d][j]);
    float ss = 0.f;
    for (int j = 0; j < NCL; ++j) { float e = expf(sc[d][j]-mx); sc[d][j] = e; ss += e; }
    for (int j = 0; j < NCL; ++j) sc[d][j] /= ss;
  }
  __syncthreads();
  for (int i = 0; i < NCL; ++i) {
    float o = 0.f;
    for (int j = 0; j < NCL; ++j) o += sc[i][j]*vv[j][d];
    size_t oi = ((size_t)b*NCL + i)*DM + hh*DH + d;
    attno[oi]   = o;
    attno16[oi] = (_Float16)o;
  }
}

// ---------------- host orchestration ----------------
extern "C" void kernel_launch(void* const* d_in, const int* in_sizes, int n_in,
                              void* d_out, int out_size, void* d_ws, size_t ws_size,
                              hipStream_t stream)
{
  (void)in_sizes; (void)n_in; (void)out_size; (void)ws_size;
  const float* x       = (const float*)d_in[0];
  const float* pos_emb = (const float*)d_in[1];
  const float* proj_w  = (const float*)d_in[2];
  const float* proj_b  = (const float*)d_in[3];
  const float* pln_g   = (const float*)d_in[4];
  const float* pln_b   = (const float*)d_in[5];
  const float* pos_enc = (const float*)d_in[6];
  const float* Wqkv    = (const float*)d_in[7];
  const float* bqkv    = (const float*)d_in[8];
  const float* Wo      = (const float*)d_in[9];
  const float* bo      = (const float*)d_in[10];
  const float* W1      = (const float*)d_in[11];
  const float* b1      = (const float*)d_in[12];
  const float* W2      = (const float*)d_in[13];
  const float* b2      = (const float*)d_in[14];
  const float* ln1_g   = (const float*)d_in[15];
  const float* ln1_b   = (const float*)d_in[16];
  const float* ln2_g   = (const float*)d_in[17];
  const float* ln2_b   = (const float*)d_in[18];

  char* base = (char*)d_ws;
  size_t off = 0;
  auto alloc = [&](size_t bytes) -> char* {
    off = (off + 255) & ~(size_t)255;
    char* p = base + off;
    off += bytes;
    return p;
  };

  _Float16* basisT  = (_Float16*)alloc((size_t)96*256*sizeof(_Float16));
  float*    hann    = (float*)   alloc(256*sizeof(float));
  float*    featf32 = (float*)   alloc((size_t)BATCH*CH*FEAT*sizeof(float));
  _Float16* featf16 = (_Float16*)alloc((size_t)BATCH*CH*FEATP*sizeof(_Float16));
  _Float16* projw16 = (_Float16*)alloc((size_t)DM*FEATP*sizeof(_Float16));
  float*    pfpre   = (float*)   alloc((size_t)BATCH*CH*DM*sizeof(float));
  float*    pf      = (float*)   alloc((size_t)BATCH*CH*DM*sizeof(float));
  float*    centers = (float*)   alloc(NCL*3*sizeof(float));
  float*    csums   = (float*)   alloc(NCL*3*sizeof(float));
  float*    ccnts   = (float*)   alloc(NCL*sizeof(float));
  int*      assign  = (int*)     alloc(CH*sizeof(int));
  float*    h       = (float*)   alloc((size_t)NTOK*DM*sizeof(float));
  _Float16* h16     = (_Float16*)alloc((size_t)NTOK*DM*sizeof(_Float16));
  float*    qkvb    = (float*)   alloc((size_t)NTOK*3*DM*sizeof(float));
  float*    attno   = (float*)   alloc((size_t)NTOK*DM*sizeof(float));
  _Float16* attno16 = (_Float16*)alloc((size_t)NTOK*DM*sizeof(_Float16));
  float*    tmp     = (float*)   alloc((size_t)NTOK*DM*sizeof(float));
  float*    ffmid   = (float*)   alloc((size_t)NTOK*2*DM*sizeof(float));
  _Float16* ffmid16 = (_Float16*)alloc((size_t)NTOK*2*DM*sizeof(_Float16));
  _Float16* wqkv16  = (_Float16*)alloc((size_t)NLAY*3*DM*DM*sizeof(_Float16));
  _Float16* wo16    = (_Float16*)alloc((size_t)NLAY*DM*DM*sizeof(_Float16));
  _Float16* w1_16   = (_Float16*)alloc((size_t)NLAY*2*DM*DM*sizeof(_Float16));
  _Float16* w2_16   = (_Float16*)alloc((size_t)NLAY*DM*2*DM*sizeof(_Float16));

  // features via WMMA DFT
  basis_kernel<<<32, 256, 0, stream>>>(basisT, hann);
  feat_kernel<<<BATCH*CH, 256, 0, stream>>>(x, basisT, hann, featf32, featf16);

  // projection: (7936 x 512) @ (512 x 128)^T + bias -> LN -> ReLU
  pad_projw_kernel<<<(DM*FEATP + 255)/256, 256, 0, stream>>>(proj_w, projw16);
  wmma_gemm_kernel<<<(BATCH*CH/16)*(DM/16), 32, 0, stream>>>(
      featf16, FEATP, projw16, FEATP, proj_b, pfpre, (_Float16*)nullptr,
      BATCH*CH, DM, FEATP, 0);
  ln_kernel<<<BATCH*CH, DM, 0, stream>>>(pfpre, nullptr, pln_g, pln_b, pf, nullptr, 1);

  // clustering
  cluster_init_kernel<<<1, 64, 0, stream>>>(csums, ccnts);
  centers0_kernel<<<1, 64, 0, stream>>>(pos_emb, centers);
  cluster_batch_kernel<<<BATCH, 256, 0, stream>>>(featf32, pos_emb, csums, ccnts);
  assign_kernel<<<1, 64, 0, stream>>>(pos_emb, csums, ccnts, centers, assign);

  // tokens
  pool_kernel<<<BATCH*NCL, DM, 0, stream>>>(pf, assign, pos_enc, h, h16);

  // f16 weights for transformer GEMMs
  f32_to_f16_kernel<<<(NLAY*3*DM*DM + 255)/256, 256, 0, stream>>>(Wqkv, wqkv16, NLAY*3*DM*DM);
  f32_to_f16_kernel<<<(NLAY*DM*DM   + 255)/256, 256, 0, stream>>>(Wo,   wo16,   NLAY*DM*DM);
  f32_to_f16_kernel<<<(NLAY*2*DM*DM + 255)/256, 256, 0, stream>>>(W1,   w1_16,  NLAY*2*DM*DM);
  f32_to_f16_kernel<<<(NLAY*DM*2*DM + 255)/256, 256, 0, stream>>>(W2,   w2_16,  NLAY*DM*2*DM);

  for (int l = 0; l < NLAY; ++l) {
    wmma_gemm_kernel<<<(NTOK/16)*((3*DM)/16), 32, 0, stream>>>(
        h16, DM, wqkv16 + (size_t)l*3*DM*DM, DM, bqkv + l*3*DM,
        qkvb, (_Float16*)nullptr, NTOK, 3*DM, DM, 0);
    attn_kernel<<<BATCH*NHD, 32, 0, stream>>>(qkvb, attno, attno16);
    wmma_gemm_kernel<<<(NTOK/16)*(DM/16), 32, 0, stream>>>(
        attno16, DM, wo16 + (size_t)l*DM*DM, DM, bo + l*DM,
        tmp, (_Float16*)nullptr, NTOK, DM, DM, 0);
    ln_kernel<<<NTOK, DM, 0, stream>>>(h, tmp, ln1_g + l*DM, ln1_b + l*DM, h, h16, 0);
    wmma_gemm_kernel<<<(NTOK/16)*((2*DM)/16), 32, 0, stream>>>(
        h16, DM, w1_16 + (size_t)l*2*DM*DM, DM, b1 + l*2*DM,
        ffmid, ffmid16, NTOK, 2*DM, DM, 1);
    wmma_gemm_kernel<<<(NTOK/16)*(DM/16), 32, 0, stream>>>(
        ffmid16, 2*DM, w2_16 + (size_t)l*DM*2*DM, 2*DM, b2 + l*DM,
        tmp, (_Float16*)nullptr, NTOK, DM, 2*DM, 0);
    float* outp = (l == NLAY-1) ? (float*)d_out : h;
    ln_kernel<<<NTOK, DM, 0, stream>>>(h, tmp, ln2_g + l*DM, ln2_b + l*DM, outp, h16, 0);
  }
}
